// TransformerDiffusion_16587163697748
// MI455X (gfx1250) — matrix operand order repaired and checked by reading, not compiled, not run on hardware
//
#include <hip/hip_runtime.h>
#include <hip/hip_bf16.h>

typedef unsigned short u16;
typedef unsigned int   u32;

#define Bn    2
#define Mn    512
#define Nv    1024
#define Dn    256
#define NHn   8
#define DHn   32
#define DFFn  1024
#define Ln    6
#define SEQn  1536
#define MASKW (SEQn / 32)   // 48 words per row

// gfx1250 async global->LDS DMA (ASYNCcnt path); fallback = sync LDS copy
#if defined(__HIP_DEVICE_COMPILE__) && __has_builtin(__builtin_amdgcn_global_load_async_to_lds_b128)
#define HAVE_ASYNC_LDS 1
#else
#define HAVE_ASYNC_LDS 0
#endif

// exact pointee type expected by the async builtin (from hipcc diagnostic):
// '__attribute__((__vector_size__(4 * sizeof(int)))) int' in AS1 / AS3
typedef int v4i __attribute__((vector_size(16)));
typedef v4i __attribute__((address_space(1))) as1_v4i;
typedef v4i __attribute__((address_space(3))) as3_v4i;

typedef __attribute__((ext_vector_type(16))) __bf16 v16bf;
typedef __attribute__((ext_vector_type(8)))  float  v8f;

union BF16x16 {
  v16bf v;
  u32   u[8];
  u16   s[16];
};

__device__ __forceinline__ u16 f2bf(float f) {
  u32 u = __float_as_uint(f);
  u32 r = u + 0x7FFFu + ((u >> 16) & 1u);
  return (u16)(r >> 16);
}

__device__ __forceinline__ void wait_asynccnt0() {
#if HAVE_ASYNC_LDS
#if __has_builtin(__builtin_amdgcn_s_wait_asynccnt)
  __builtin_amdgcn_s_wait_asynccnt(0);
#else
  asm volatile("s_wait_asynccnt 0x0" ::: "memory");
#endif
#endif
}

// ---------------------------------------------------------------------------
// syndrome: synd[b][m] = parity over n of H[m][n] * (r<0)
// ---------------------------------------------------------------------------
__global__ void syndrome_kernel(const float* __restrict__ r,
                                const int* __restrict__ H,
                                float* __restrict__ synd) {
  int idx = blockIdx.x * blockDim.x + threadIdx.x;
  if (idx >= Bn * Mn) return;
  int b = idx / Mn, m = idx % Mn;
  int p = 0;
  for (int n = 0; n < Nv; ++n)
    if (H[m * Nv + n]) p ^= (r[b * Nv + n] < 0.0f) ? 1 : 0;
  synd[idx] = (float)p;
}

// ---------------------------------------------------------------------------
// x[b][s][d] = src_embed[s][d] * node[b][s] * time_table[t[b]][d]
// ---------------------------------------------------------------------------
__global__ void build_x(const float* __restrict__ r, const float* __restrict__ synd,
                        const int* __restrict__ t, const float* __restrict__ se,
                        const float* __restrict__ tt,
                        float* __restrict__ x, u16* __restrict__ xbf) {
  int row = blockIdx.x;              // b*SEQ + s
  int b = row / SEQn, s = row % SEQn;
  int d = threadIdx.x;
  float node = (s < Nv) ? fabsf(r[b * Nv + s]) : synd[b * Mn + (s - Nv)];
  float v = se[s * Dn + d] * node * tt[t[b] * Dn + d];
  size_t i = (size_t)row * Dn + d;
  x[i] = v;
  xbf[i] = f2bf(v);
}

// ---------------------------------------------------------------------------
// Tanner adjacency as bit mask: bit k of maskBits[q*MASKW + k/32]
// ---------------------------------------------------------------------------
__global__ void build_mask(const int* __restrict__ H, u32* __restrict__ mb) {
  int idx = blockIdx.x * blockDim.x + threadIdx.x;
  if (idx >= SEQn * MASKW) return;
  int q = idx / MASKW, w = idx % MASKW;
  u32 bits = 0;
  for (int i = 0; i < 32; ++i) {
    int k = w * 32 + i;
    bool adj = (q == k);
    if (q < Nv && k >= Nv) adj |= (H[(k - Nv) * Nv + q] != 0);
    if (q >= Nv && k < Nv) adj |= (H[(q - Nv) * Nv + k] != 0);
    if (adj) bits |= (1u << i);
  }
  mb[idx] = bits;
}

// ---------------------------------------------------------------------------
// W (K x N) f32 row-major  ->  WT (N x K) bf16  (contiguous-K B fragments)
// ---------------------------------------------------------------------------
__global__ void transpose_w(const float* __restrict__ W, u16* __restrict__ WT,
                            int K, int N) {
  int idx = blockIdx.x * blockDim.x + threadIdx.x;
  if (idx >= K * N) return;
  int k = idx / N, n = idx % N;
  WT[(size_t)n * K + k] = f2bf(W[idx]);
}

// ---------------------------------------------------------------------------
// vbf[(b*SEQ+s)*D + d] -> vt[(b*D + d)*SEQ + s]  (per-(b,h,dh) key rows)
// ---------------------------------------------------------------------------
__global__ void transpose_v(const u16* __restrict__ vbf, u16* __restrict__ vt) {
  int idx = blockIdx.x * blockDim.x + threadIdx.x;   // over B*SEQ*D
  int d = idx % Dn;
  int rs = idx / Dn;
  int s = rs % SEQn, b = rs / SEQn;
  vt[((size_t)b * Dn + d) * SEQn + s] = vbf[idx];
}

// ---------------------------------------------------------------------------
// C = A(R x K, bf16) @ BT^T (BT is N x K bf16) + bias; optional relu.
// One 16x16 tile per wave, 8 waves/block share one 16-col B tile which is
// staged into LDS once via async global->LDS DMA (ASYNCcnt).
// ---------------------------------------------------------------------------
__global__ __launch_bounds__(256)
void gemm_bf16(const u16* __restrict__ A, const u16* __restrict__ BT,
               const float* __restrict__ bias, float* __restrict__ Cf,
               u16* __restrict__ Cbf, int R, int K, int Ncol, int relu) {
  __shared__ u16 ldsB[16 * DFFn];                 // 32KB max (K<=1024)
  int lane = threadIdx.x & 31;
  int wv = threadIdx.x >> 5;
  int rowTile = blockIdx.y * 8 + wv;
  int colTile = blockIdx.x;
  int row0 = rowTile * 16, col0 = colTile * 16;
  int lh = lane >> 4, ln = lane & 15;

  // ---- cooperative stage of the 16 x K B tile (contiguous in BT) ----
  const u16* bsrc = BT + (size_t)col0 * K;
  for (int i = threadIdx.x * 8; i < 16 * K; i += 256 * 8) {
#if HAVE_ASYNC_LDS
    __builtin_amdgcn_global_load_async_to_lds_b128(
        (as1_v4i*)(bsrc + i), (as3_v4i*)(ldsB + i), 0, 0);
#else
    *(uint4*)(ldsB + i) = *(const uint4*)(bsrc + i);
#endif
  }
  wait_asynccnt0();
  __syncthreads();

  v8f acc = {};
  const u16* arow = A + (size_t)(row0 + ln) * K + lh * 16;
  const u16* bcol = ldsB + ln * K + lh * 16;      // LDS, contiguous K
  for (int k0 = 0; k0 < K; k0 += 32) {
    BF16x16 a, b;
#pragma unroll
    for (int v = 0; v < 8; ++v) {
      int kb = v * 2;
      a.u[v] = *(const u32*)(arow + k0 + kb);
    }
    *(uint4*)&b.u[0] = *(const uint4*)(bcol + k0);      // ds_load_b128
    *(uint4*)&b.u[4] = *(const uint4*)(bcol + k0 + 8);
    if (k0 + 32 < K)
      __builtin_prefetch(arow + k0 + 32, 0, 1);         // global_prefetch_b8
    acc = __builtin_amdgcn_wmma_f32_16x16x32_bf16(
        false, a.v, false, b.v, (short)0, acc, false, false);
  }
#pragma unroll
  for (int j = 0; j < 8; ++j) {
    int rr = row0 + lh * 8 + j;
    int cc = col0 + ln;
    float v = acc[j] + bias[cc];
    if (relu) v = fmaxf(v, 0.0f);
    size_t i = (size_t)rr * Ncol + cc;
    if (Cf)  Cf[i] = v;
    if (Cbf) Cbf[i] = f2bf(v);
  }
}

// ---------------------------------------------------------------------------
// Flash attention per (b, h, 16-query tile). Q@K^T and P@V as WMMA.
// V is consumed from the transposed vt layout -> contiguous b128 fragments.
// ---------------------------------------------------------------------------
__global__ __launch_bounds__(256)
void attention_kernel(const u16* __restrict__ Qb, const u16* __restrict__ Kb,
                      const u16* __restrict__ vt, const u32* __restrict__ mb,
                      u16* __restrict__ Ob) {
  __shared__ u16 probs[8][16 * 32];
  int lane = threadIdx.x & 31;
  int wv = threadIdx.x >> 5;
  int qTile = blockIdx.x * 8 + wv;
  int h = blockIdx.y;
  int b = blockIdx.z;
  int lh = lane >> 4, ln = lane & 15;
  const float scale = 0.17677669529663687f;     // 1/sqrt(DH)
  size_t base = (size_t)b * SEQn * Dn;

  // Q A-fragment (16 rows x DH=32)
  BF16x16 qf;
  const u16* qrow = Qb + base + (size_t)(qTile * 16 + ln) * Dn + h * DHn + lh * 16;
#pragma unroll
  for (int v = 0; v < 8; ++v)
    qf.u[v] = *(const u32*)(qrow + v * 2);

  // V row bases in the transposed layout: dh = ln (o0) / 16+ln (o1)
  const u16* vrow0 = vt + ((size_t)b * Dn + h * DHn + ln) * SEQn + lh * 16;
  const u16* vrow1 = vt + ((size_t)b * Dn + h * DHn + 16 + ln) * SEQn + lh * 16;

  float mrow[8], lrow[8];
  v8f o0 = {}, o1 = {};
#pragma unroll
  for (int j = 0; j < 8; ++j) { mrow[j] = -3.0e38f; lrow[j] = 0.0f; }

  for (int kc = 0; kc < SEQn; kc += 32) {
    v8f z = {};
    // scores for keys [kc, kc+16) and [kc+16, kc+32)
    BF16x16 kf0, kf1;
    const u16* kr0 = Kb + base + (size_t)(kc + ln) * Dn + h * DHn + lh * 16;
    const u16* kr1 = kr0 + (size_t)16 * Dn;
    *(uint4*)&kf0.u[0] = *(const uint4*)(kr0);
    *(uint4*)&kf0.u[4] = *(const uint4*)(kr0 + 8);
    *(uint4*)&kf1.u[0] = *(const uint4*)(kr1);
    *(uint4*)&kf1.u[4] = *(const uint4*)(kr1 + 8);
    __builtin_prefetch(kr0 + 32 * Dn, 0, 1);
    v8f s0 = __builtin_amdgcn_wmma_f32_16x16x32_bf16(
        false, qf.v, false, kf0.v, (short)0, z, false, false);
    v8f s1 = __builtin_amdgcn_wmma_f32_16x16x32_bf16(
        false, qf.v, false, kf1.v, (short)0, z, false, false);

#pragma unroll
    for (int j = 0; j < 8; ++j) {
      int qg = qTile * 16 + lh * 8 + j;
      int k0g = kc + ln, k1g = kc + 16 + ln;
      u32 w0 = mb[qg * MASKW + (k0g >> 5)];
      u32 w1 = mb[qg * MASKW + (k1g >> 5)];
      float a0 = s0[j] * scale + (((w0 >> (k0g & 31)) & 1u) ? 0.0f : -1e9f);
      float a1 = s1[j] * scale + (((w1 >> (k1g & 31)) & 1u) ? 0.0f : -1e9f);
      // row max across the 16 lanes of this half-wave
      float cm = fmaxf(a0, a1);
      for (int msk = 1; msk < 16; msk <<= 1) cm = fmaxf(cm, __shfl_xor(cm, msk, 32));
      float mn = fmaxf(mrow[j], cm);
      float corr = __expf(mrow[j] - mn);
      float p0 = __expf(a0 - mn);
      float p1 = __expf(a1 - mn);
      float rs = p0 + p1;
      for (int msk = 1; msk < 16; msk <<= 1) rs += __shfl_xor(rs, msk, 32);
      lrow[j] = lrow[j] * corr + rs;
      o0[j] *= corr;
      o1[j] *= corr;
      mrow[j] = mn;
      probs[wv][(lh * 8 + j) * 32 + ln]      = f2bf(p0);
      probs[wv][(lh * 8 + j) * 32 + 16 + ln] = f2bf(p1);
    }
    // LDS in-order per wave; make the transposed re-read explicit
    asm volatile("s_wait_dscnt 0x0" ::: "memory");

    // P A-fragment (16 x 32) from the per-wave LDS tile
    BF16x16 pf;
#pragma unroll
    for (int v = 0; v < 8; ++v) {
      int kb = lh * 16 + v * 2;
      pf.u[v] = *(const u32*)&probs[wv][ln * 32 + kb];
    }
    // V B-fragments (32 keys x 16 dh) as contiguous b128 from vt
    BF16x16 vf0, vf1;
    *(uint4*)&vf0.u[0] = *(const uint4*)(vrow0 + kc);
    *(uint4*)&vf0.u[4] = *(const uint4*)(vrow0 + kc + 8);
    *(uint4*)&vf1.u[0] = *(const uint4*)(vrow1 + kc);
    *(uint4*)&vf1.u[4] = *(const uint4*)(vrow1 + kc + 8);
    o0 = __builtin_amdgcn_wmma_f32_16x16x32_bf16(
        false, pf.v, false, vf0.v, (short)0, o0, false, false);
    o1 = __builtin_amdgcn_wmma_f32_16x16x32_bf16(
        false, pf.v, false, vf1.v, (short)0, o1, false, false);
  }

#pragma unroll
  for (int j = 0; j < 8; ++j) {
    int tok = qTile * 16 + lh * 8 + j;
    float inv = 1.0f / lrow[j];
    size_t oidx = base + (size_t)tok * Dn + h * DHn;
    Ob[oidx + ln]      = f2bf(o0[j] * inv);
    Ob[oidx + 16 + ln] = f2bf(o1[j] * inv);
  }
}

// ---------------------------------------------------------------------------
// x = LayerNorm(xin + y) * g + b   (one row per block, D=256 threads)
// ---------------------------------------------------------------------------
__global__ __launch_bounds__(256)
void add_ln(const float* __restrict__ xin, const float* __restrict__ y,
            const float* __restrict__ g, const float* __restrict__ bt,
            float* __restrict__ xout, u16* __restrict__ xbf) {
  __shared__ float sh[2][8];
  int row = blockIdx.x, d = threadIdx.x;
  size_t i = (size_t)row * Dn + d;
  float v = xin[i] + y[i];
  float s = v, s2 = v * v;
  for (int msk = 16; msk >= 1; msk >>= 1) {
    s += __shfl_xor(s, msk, 32);
    s2 += __shfl_xor(s2, msk, 32);
  }
  int wv = threadIdx.x >> 5, lane = threadIdx.x & 31;
  if (lane == 0) { sh[0][wv] = s; sh[1][wv] = s2; }
  __syncthreads();
  float su = 0.0f, sq = 0.0f;
#pragma unroll
  for (int k = 0; k < 8; ++k) { su += sh[0][k]; sq += sh[1][k]; }
  float mu = su / Dn;
  float var = sq / Dn - mu * mu;
  float inv = rsqrtf(var + 1e-6f);
  float o = g[d] * (v - mu) * inv + bt[d];
  xout[i] = o;
  xbf[i] = f2bf(o);
}

// ---------------------------------------------------------------------------
// out[b][n] = to_n_w * (x[b,n,:] . fc_w + fc_b) + to_n_b   (n < Nv)
// ---------------------------------------------------------------------------
__global__ __launch_bounds__(256)
void head_kernel(const float* __restrict__ x, const float* __restrict__ fcw,
                 const float* __restrict__ fcb, const float* __restrict__ tnw,
                 const float* __restrict__ tnb, float* __restrict__ out) {
  __shared__ float sh[8];
  int b = blockIdx.x / Nv, n = blockIdx.x % Nv;
  int d = threadIdx.x;
  float p = x[((size_t)b * SEQn + n) * Dn + d] * fcw[d];
  for (int msk = 16; msk >= 1; msk >>= 1) p += __shfl_xor(p, msk, 32);
  int wv = threadIdx.x >> 5, lane = threadIdx.x & 31;
  if (lane == 0) sh[wv] = p;
  __syncthreads();
  if (threadIdx.x == 0) {
    float dot = 0.0f;
#pragma unroll
    for (int k = 0; k < 8; ++k) dot += sh[k];
    out[b * Nv + n] = tnw[0] * (dot + fcb[0]) + tnb[0];
  }
}

// ---------------------------------------------------------------------------
extern "C" void kernel_launch(void* const* d_in, const int* in_sizes, int n_in,
                              void* d_out, int out_size, void* d_ws, size_t ws_size,
                              hipStream_t stream) {
  (void)in_sizes; (void)n_in; (void)out_size; (void)ws_size;
  const float* r_t  = (const float*)d_in[0];
  const int*   tt_i = (const int*)d_in[1];
  const int*   H    = (const int*)d_in[2];
  const float* se   = (const float*)d_in[3];
  const float* timt = (const float*)d_in[4];
  const float* Wq = (const float*)d_in[5];  const float* bq = (const float*)d_in[6];
  const float* Wk = (const float*)d_in[7];  const float* bk = (const float*)d_in[8];
  const float* Wv = (const float*)d_in[9];  const float* bv = (const float*)d_in[10];
  const float* Wo = (const float*)d_in[11]; const float* bo = (const float*)d_in[12];
  const float* ln1g = (const float*)d_in[13]; const float* ln1b = (const float*)d_in[14];
  const float* W1 = (const float*)d_in[15]; const float* b1 = (const float*)d_in[16];
  const float* W2 = (const float*)d_in[17]; const float* b2 = (const float*)d_in[18];
  const float* ln2g = (const float*)d_in[19]; const float* ln2b = (const float*)d_in[20];
  const float* fcw = (const float*)d_in[21]; const float* fcb = (const float*)d_in[22];
  const float* tnw = (const float*)d_in[23]; const float* tnb = (const float*)d_in[24];
  float* out = (float*)d_out;

  // bump allocator on workspace
  size_t off = 0;
  auto alloc = [&](size_t bytes) -> void* {
    off = (off + 255) & ~(size_t)255;
    void* p = (char*)d_ws + off;
    off += bytes;
    return p;
  };
  const int R = Bn * SEQn;                       // 3072 token rows
  float* synd   = (float*)alloc(Bn * Mn * sizeof(float));
  float* x      = (float*)alloc((size_t)R * Dn * sizeof(float));
  u16*   xbf    = (u16*)  alloc((size_t)R * Dn * sizeof(u16));
  u32*   mbits  = (u32*)  alloc((size_t)SEQn * MASKW * sizeof(u32));
  u16*   qbf    = (u16*)  alloc((size_t)R * Dn * sizeof(u16));
  u16*   kbf    = (u16*)  alloc((size_t)R * Dn * sizeof(u16));
  u16*   vbf    = (u16*)  alloc((size_t)R * Dn * sizeof(u16));
  u16*   vt     = (u16*)  alloc((size_t)R * Dn * sizeof(u16));
  u16*   abf    = (u16*)  alloc((size_t)R * Dn * sizeof(u16));
  float* yf     = (float*)alloc((size_t)R * Dn * sizeof(float));
  u16*   hbf    = (u16*)  alloc((size_t)R * DFFn * sizeof(u16));
  u16*   wqT    = (u16*)  alloc((size_t)Ln * Dn * Dn * sizeof(u16));
  u16*   wkT    = (u16*)  alloc((size_t)Ln * Dn * Dn * sizeof(u16));
  u16*   wvT    = (u16*)  alloc((size_t)Ln * Dn * Dn * sizeof(u16));
  u16*   woT    = (u16*)  alloc((size_t)Ln * Dn * Dn * sizeof(u16));
  u16*   w1T    = (u16*)  alloc((size_t)Ln * Dn * DFFn * sizeof(u16));
  u16*   w2T    = (u16*)  alloc((size_t)Ln * DFFn * Dn * sizeof(u16));

  // --- setup ---
  syndrome_kernel<<<(Bn * Mn + 255) / 256, 256, 0, stream>>>(r_t, H, synd);
  build_x<<<R, Dn, 0, stream>>>(r_t, synd, tt_i, se, timt, x, xbf);
  build_mask<<<(SEQn * MASKW + 255) / 256, 256, 0, stream>>>(H, mbits);
  for (int l = 0; l < Ln; ++l) {
    int ddBlocks = (Dn * Dn + 255) / 256;
    int dfBlocks = (Dn * DFFn + 255) / 256;
    transpose_w<<<ddBlocks, 256, 0, stream>>>(Wq + (size_t)l * Dn * Dn, wqT + (size_t)l * Dn * Dn, Dn, Dn);
    transpose_w<<<ddBlocks, 256, 0, stream>>>(Wk + (size_t)l * Dn * Dn, wkT + (size_t)l * Dn * Dn, Dn, Dn);
    transpose_w<<<ddBlocks, 256, 0, stream>>>(Wv + (size_t)l * Dn * Dn, wvT + (size_t)l * Dn * Dn, Dn, Dn);
    transpose_w<<<ddBlocks, 256, 0, stream>>>(Wo + (size_t)l * Dn * Dn, woT + (size_t)l * Dn * Dn, Dn, Dn);
    transpose_w<<<dfBlocks, 256, 0, stream>>>(W1 + (size_t)l * Dn * DFFn, w1T + (size_t)l * Dn * DFFn, Dn, DFFn);
    transpose_w<<<dfBlocks, 256, 0, stream>>>(W2 + (size_t)l * DFFn * Dn, w2T + (size_t)l * DFFn * Dn, DFFn, Dn);
  }

  dim3 blk(256);
  dim3 gProj(Dn / 16, R / 128);                  // (16, 24)
  dim3 gFfn1(DFFn / 16, R / 128);                // (64, 24)
  dim3 gAttn(SEQn / 128, NHn, Bn);               // (12, 8, 2)

  for (int l = 0; l < Ln; ++l) {
    const u16* wq = wqT + (size_t)l * Dn * Dn;
    const u16* wk = wkT + (size_t)l * Dn * Dn;
    const u16* wv = wvT + (size_t)l * Dn * Dn;
    const u16* wo = woT + (size_t)l * Dn * Dn;
    const u16* w1 = w1T + (size_t)l * Dn * DFFn;
    const u16* w2 = w2T + (size_t)l * DFFn * Dn;
    // Q, K, V projections (bf16 outputs feed WMMA attention)
    gemm_bf16<<<gProj, blk, 0, stream>>>(xbf, wq, bq + (size_t)l * Dn, nullptr, qbf, R, Dn, Dn, 0);
    gemm_bf16<<<gProj, blk, 0, stream>>>(xbf, wk, bk + (size_t)l * Dn, nullptr, kbf, R, Dn, Dn, 0);
    gemm_bf16<<<gProj, blk, 0, stream>>>(xbf, wv, bv + (size_t)l * Dn, nullptr, vbf, R, Dn, Dn, 0);
    transpose_v<<<(R * Dn) / 256, blk, 0, stream>>>(vbf, vt);
    // attention
    attention_kernel<<<gAttn, blk, 0, stream>>>(qbf, kbf, vt, mbits, abf);
    // output projection + residual LN
    gemm_bf16<<<gProj, blk, 0, stream>>>(abf, wo, bo + (size_t)l * Dn, yf, nullptr, R, Dn, Dn, 0);
    add_ln<<<R, Dn, 0, stream>>>(x, yf, ln1g + (size_t)l * Dn, ln1b + (size_t)l * Dn, x, xbf);
    // FFN
    gemm_bf16<<<gFfn1, blk, 0, stream>>>(xbf, w1, b1 + (size_t)l * DFFn, nullptr, hbf, R, Dn, DFFn, 1);
    gemm_bf16<<<gProj, blk, 0, stream>>>(hbf, w2, b2 + (size_t)l * Dn, yf, nullptr, R, DFFn, Dn, 0);
    add_ln<<<R, Dn, 0, stream>>>(x, yf, ln2g + (size_t)l * Dn, ln2b + (size_t)l * Dn, x, xbf);
  }
  // head over variable nodes
  head_kernel<<<Bn * Nv, Dn, 0, stream>>>(x, fcw, fcb, tnw, tnb, out);
}
